// HeteroGATModel_71622874628356
// MI455X (gfx1250) — matrix operand order repaired and checked by reading, not compiled, not run on hardware
//
#include <hip/hip_runtime.h>
#include <math.h>

typedef __bf16 bf16_t;
typedef bf16_t v16bf __attribute__((ext_vector_type(16)));
typedef float  v8f   __attribute__((ext_vector_type(8)));

#define NPAPER 100000
#define NAUTH  50000
#define HID    256
#define NEDGE  250000
#define NH     4
#define NC     64

// ---------------- device helpers ----------------

__device__ __forceinline__ void atomicMaxF(float* addr, float v) {
  // sign-trick float max: avoids CAS loops entirely
  if (v >= 0.0f) atomicMax((int*)addr, __float_as_int(v));
  else           atomicMin((unsigned int*)addr, __float_as_uint(v));
}

// A fragment (16x32 bf16 slice for one row): element e -> K = (e>>3)*16 + hi*8 + (e&7)
__device__ __forceinline__ v16bf load_a_frag(const float* __restrict__ arow) {
  float4 a0 = *(const float4*)(arow);
  float4 a1 = *(const float4*)(arow + 4);
  float4 a2 = *(const float4*)(arow + 16);
  float4 a3 = *(const float4*)(arow + 20);
  v16bf a;
  a[0]  = (bf16_t)a0.x; a[1]  = (bf16_t)a0.y; a[2]  = (bf16_t)a0.z; a[3]  = (bf16_t)a0.w;
  a[4]  = (bf16_t)a1.x; a[5]  = (bf16_t)a1.y; a[6]  = (bf16_t)a1.z; a[7]  = (bf16_t)a1.w;
  a[8]  = (bf16_t)a2.x; a[9]  = (bf16_t)a2.y; a[10] = (bf16_t)a2.z; a[11] = (bf16_t)a2.w;
  a[12] = (bf16_t)a3.x; a[13] = (bf16_t)a3.y; a[14] = (bf16_t)a3.z; a[15] = (bf16_t)a3.w;
  return a;
}

// ---------------- W pre-pack: fp32 W[K,N] -> bf16 Wp[(K/32)][N][32] ----------------
__global__ void k_pack_w(const float* __restrict__ W, bf16_t* __restrict__ Wp,
                         int K, int N) {
  int idx = blockIdx.x * blockDim.x + threadIdx.x;
  if (idx >= K * N) return;
  int k = idx / N, n = idx - k * N;
  int blk = k >> 5, kk = k & 31;
  Wp[((size_t)blk * N + n) * 32 + kk] = (bf16_t)W[(size_t)k * N + n];
}

// ---------------- WMMA GEMM: C[M,N] = act(A[M,K] @ W[K,N] + bias) ----------------
// One wave computes a 32x64 tile (8 accumulators, 2 A-fragments sharing each B
// fragment). K is a compile-time constant so the k-loop fully unrolls and the
// scheduler can pipeline loads over WMMAs. M % 16 == 0; if the tail row-group
// has only 16 valid rows, upper-half A reads are clamped and stores predicated
// (wave-uniform, so EXEC is all-ones around every WMMA).
template <int K>
__global__ void k_gemm_wmma(const float* __restrict__ A, const bf16_t* __restrict__ Wp,
                            float* __restrict__ C, const float* __restrict__ bias,
                            int M, int N, int relu) {
  int gtid   = blockIdx.x * blockDim.x + threadIdx.x;
  int wave   = gtid >> 5;
  int lane   = threadIdx.x & 31;
  int tilesN = N >> 6;
  int m0 = (wave / tilesN) << 5;
  int n0 = (wave % tilesN) << 6;
  if (m0 >= M) return;                 // whole wave exits uniformly
  int mrow = lane & 15;
  int hi   = lane >> 4;                // lane half select
  bool hasHi = (m0 + 16) < M;          // wave-uniform
  int rowL = m0 + mrow;
  int rowH = hasHi ? (m0 + 16 + mrow) : rowL;   // clamp to valid memory

  v8f accL[4] = {}, accH[4] = {};
  #pragma unroll
  for (int k0 = 0; k0 < K; k0 += 32) {
    v16bf aL = load_a_frag(A + (size_t)rowL * K + k0 + hi * 8);
    v16bf aH = load_a_frag(A + (size_t)rowH * K + k0 + hi * 8);
    // B fragment: lane -> column n = n0 + t*16 + mrow, element e -> K = k0 + (e>>3)*16 + hi*8 + (e&7)
    const bf16_t* wbase = Wp + ((size_t)(k0 >> 5) * N + n0 + mrow) * 32 + hi * 8;
    #pragma unroll
    for (int t = 0; t < 4; ++t) {
      v16bf b;
      uint4* bb = (uint4*)&b;
      const uint4* wsrc = (const uint4*)(wbase + (size_t)t * 512);
      bb[0] = wsrc[0];
      bb[1] = wsrc[2];
      accL[t] = __builtin_amdgcn_wmma_f32_16x16x32_bf16(
          false, aL, false, b, (short)0, accL[t], false, false);
      accH[t] = __builtin_amdgcn_wmma_f32_16x16x32_bf16(
          false, aH, false, b, (short)0, accH[t], false, false);
    }
  }
  // D layout: VGPR r -> M = base + hi*8 + r, N = n0 + t*16 + mrow
  #pragma unroll
  for (int t = 0; t < 4; ++t) {
    int col = n0 + t * 16 + mrow;
    float bv = bias ? bias[col] : 0.0f;
    #pragma unroll
    for (int r = 0; r < 8; ++r) {
      int row = m0 + hi * 8 + r;
      float v = accL[t][r] + bv;
      if (relu) v = fmaxf(v, 0.0f);
      C[(size_t)row * N + col] = v;
    }
    if (hasHi) {
      #pragma unroll
      for (int r = 0; r < 8; ++r) {
        int row = m0 + 16 + hi * 8 + r;
        float v = accH[t][r] + bv;
        if (relu) v = fmaxf(v, 0.0f);
        C[(size_t)row * N + col] = v;
      }
    }
  }
}

// ---------------- small kernels ----------------

// v[k,h] = sum_c W[k, h*64+c] * att[h*64+c]   (folds att into the dst/src linear)
__global__ void k_fold(const float* __restrict__ Wl, const float* __restrict__ att,
                       float* __restrict__ v, int K) {
  int idx = blockIdx.x * blockDim.x + threadIdx.x;
  if (idx >= K * NH) return;
  int k = idx >> 2, h = idx & 3;
  const float* wrow = Wl + (size_t)k * HID + h * NC;
  const float* arow = att + h * NC;
  float s = 0.f;
  for (int c = 0; c < NC; ++c) s += wrow[c] * arow[c];
  v[idx] = s;
}

// al[n,h] = sum_k X[n,k] * v[k,h]
__global__ void k_alpha_lin(const float* __restrict__ X, const float* __restrict__ v,
                            float* __restrict__ al, int Nn) {
  int idx = blockIdx.x * blockDim.x + threadIdx.x;
  if (idx >= Nn * NH) return;
  int n = idx >> 2, h = idx & 3;
  const float* xr = X + (size_t)n * HID;
  float s = 0.f;
  for (int k = 0; k < HID; ++k) s += xr[k] * v[k * NH + h];
  al[idx] = s;
}

__global__ void k_fill(float* p, float v, int n) {
  int i = blockIdx.x * blockDim.x + threadIdx.x;
  if (i < n) p[i] = v;
}

__global__ void k_init_bias(float* out, const float* __restrict__ b0,
                            const float* __restrict__ b1, int n) {
  int i = blockIdx.x * blockDim.x + threadIdx.x;
  if (i >= n) return;
  int c = i & (HID - 1);
  float v = b0[c];
  if (b1) v += b1[c];
  out[i] = v;
}

// pass 1: logit = leaky_relu(al_s[src] + al_d[dst]); segment max over dst
__global__ void k_edge_logit(const int* __restrict__ src, const int* __restrict__ dst,
                             const float* __restrict__ als, const float* __restrict__ ald,
                             float* __restrict__ elog, float* __restrict__ mb, int E) {
  int idx = blockIdx.x * blockDim.x + threadIdx.x;
  if (idx >= E * NH) return;
  int e = idx >> 2, h = idx & 3;
  int s = src[e], d = dst[e];
  float l = als[s * NH + h] + ald[d * NH + h];
  l = (l > 0.f) ? l : 0.2f * l;
  elog[idx] = l;
  atomicMaxF(&mb[d * NH + h], l);
}

// pass 2: p = exp(logit - m[dst]); segment sum over dst
__global__ void k_edge_exp(const int* __restrict__ dst, float* __restrict__ elog,
                           const float* __restrict__ mb, float* __restrict__ db, int E) {
  int idx = blockIdx.x * blockDim.x + threadIdx.x;
  if (idx >= E * NH) return;
  int e = idx >> 2, h = idx & 3;
  int d = dst[e];
  float p = __expf(elog[idx] - mb[d * NH + h]);
  elog[idx] = p;
  atomicAdd(&db[d * NH + h], p);
}

// pass 3: out[dst] += (p/denom[dst]) * hs[src]   (thread = one (edge, channel))
__global__ void k_edge_scatter(const int* __restrict__ src, const int* __restrict__ dst,
                               const float* __restrict__ elog, const float* __restrict__ db,
                               const float* __restrict__ hs, float* __restrict__ out, int E) {
  long idx = (long)blockIdx.x * blockDim.x + threadIdx.x;
  if (idx >= (long)E * HID) return;
  int e = (int)(idx >> 8);
  int c = (int)(idx & (HID - 1));
  int h = c >> 6;
  int s = src[e], d = dst[e];
  float alpha = elog[e * NH + h] / db[d * NH + h];
  atomicAdd(&out[(size_t)d * HID + c], alpha * hs[(size_t)s * HID + c]);
}

__global__ void k_elu(const float* __restrict__ in, float* __restrict__ out, int n) {
  int i = blockIdx.x * blockDim.x + threadIdx.x;
  if (i >= n) return;
  float x = in[i];
  out[i] = (x > 0.f) ? x : expm1f(x);
}

// ---------------- host orchestration ----------------

extern "C" void kernel_launch(void* const* d_in, const int* in_sizes, int n_in,
                              void* d_out, int out_size, void* d_ws, size_t ws_size,
                              hipStream_t stream) {
  const float* x_paper  = (const float*)d_in[0];
  const float* x_author = (const float*)d_in[1];
  const float* W_in_p   = (const float*)d_in[2];
  const float* b_in_p   = (const float*)d_in[3];
  const float* W_in_a   = (const float*)d_in[4];
  const float* b_in_a   = (const float*)d_in[5];
  const float* lin_src  = (const float*)d_in[6];
  const float* lin_dst  = (const float*)d_in[7];
  const float* att_src  = (const float*)d_in[8];
  const float* att_dst  = (const float*)d_in[9];
  const float* bias_gat = (const float*)d_in[10];
  const int* w_src  = (const int*)d_in[11];
  const int* w_dst  = (const int*)d_in[12];
  const int* c_src  = (const int*)d_in[13];
  const int* c_dst  = (const int*)d_in[14];
  const int* wbk_src = (const int*)d_in[15];
  const int* wbk_dst = (const int*)d_in[16];

  float* ws = (float*)d_ws;
  size_t o = 0;
  float* xp   = ws + o; o += (size_t)NPAPER * HID;   // current paper features
  float* xa   = ws + o; o += (size_t)NAUTH  * HID;   // current author features
  float* outp = ws + o; o += (size_t)NPAPER * HID;   // paper accumulator
  float* outa = ws + o; o += (size_t)NAUTH  * HID;   // author accumulator
  float* hsb  = ws + o; o += (size_t)NPAPER * HID;   // hs buffer (max Ns)
  float* als  = ws + o; o += (size_t)NPAPER * NH;
  float* ald  = ws + o; o += (size_t)NPAPER * NH;
  float* mb   = ws + o; o += (size_t)NPAPER * NH;
  float* db   = ws + o; o += (size_t)NPAPER * NH;
  float* elg  = ws + o; o += (size_t)NEDGE  * NH;
  float* vsb  = ws + o; o += (size_t)HID * NH;
  float* vdb  = ws + o; o += (size_t)HID * NH;
  bf16_t* wpk = (bf16_t*)(ws + o); o += (size_t)HID * HID / 2 + 64;  // packed bf16 weights

  auto cdiv = [](long a, long b) { return (int)((a + b - 1) / b); };
  // pack W then run the WMMA GEMM (stream-ordered, so the dependency is safe)
  auto gemm = [&](const float* A, const float* W, float* C, const float* bias,
                  int M, int K, int relu) {
    k_pack_w<<<cdiv((long)K * HID, 256), 256, 0, stream>>>(W, wpk, K, HID);
    int waves = cdiv(M, 32) * (HID >> 6);
    int blocks = cdiv((long)waves * 32, 256);
    if (K == 128)
      k_gemm_wmma<128><<<blocks, 256, 0, stream>>>(A, wpk, C, bias, M, HID, relu);
    else
      k_gemm_wmma<256><<<blocks, 256, 0, stream>>>(A, wpk, C, bias, M, HID, relu);
  };

  // input projections (K = 128, ReLU)
  gemm(x_paper,  W_in_p, xp, b_in_p, NPAPER, 128, 1);
  gemm(x_author, W_in_a, xa, b_in_a, NAUTH,  128, 1);

  auto run_gat = [&](const float* xsrc, int Ns, const float* xdst, int Nd,
                     int l, int et, const int* es, const int* ed, float* outb) {
    const float* Wls = lin_src + (size_t)(l * 3 + et) * HID * HID;
    const float* Wld = lin_dst + (size_t)(l * 3 + et) * HID * HID;
    const float* as_ = att_src + (size_t)(l * 3 + et) * HID;
    const float* ad_ = att_dst + (size_t)(l * 3 + et) * HID;
    gemm(xsrc, Wls, hsb, nullptr, Ns, HID, 0);                         // hs (WMMA)
    k_fold<<<cdiv(HID * NH, 256), 256, 0, stream>>>(Wls, as_, vsb, HID);
    k_fold<<<cdiv(HID * NH, 256), 256, 0, stream>>>(Wld, ad_, vdb, HID);
    k_alpha_lin<<<cdiv((long)Ns * NH, 256), 256, 0, stream>>>(xsrc, vsb, als, Ns);
    k_alpha_lin<<<cdiv((long)Nd * NH, 256), 256, 0, stream>>>(xdst, vdb, ald, Nd);
    k_fill<<<cdiv((long)Nd * NH, 256), 256, 0, stream>>>(mb, -INFINITY, Nd * NH);
    k_fill<<<cdiv((long)Nd * NH, 256), 256, 0, stream>>>(db, 0.0f, Nd * NH);
    k_edge_logit<<<cdiv((long)NEDGE * NH, 256), 256, 0, stream>>>(es, ed, als, ald, elg, mb, NEDGE);
    k_edge_exp<<<cdiv((long)NEDGE * NH, 256), 256, 0, stream>>>(ed, elg, mb, db, NEDGE);
    k_edge_scatter<<<cdiv((long)NEDGE * HID, 256), 256, 0, stream>>>(es, ed, elg, db, hsb, outb, NEDGE);
  };

  for (int l = 0; l < 2; ++l) {
    const float* b0 = bias_gat + (size_t)(l * 3 + 0) * HID;
    const float* b1 = bias_gat + (size_t)(l * 3 + 1) * HID;
    const float* b2 = bias_gat + (size_t)(l * 3 + 2) * HID;
    k_init_bias<<<cdiv((long)NPAPER * HID, 256), 256, 0, stream>>>(outp, b0, b1, NPAPER * HID);
    k_init_bias<<<cdiv((long)NAUTH * HID, 256), 256, 0, stream>>>(outa, b2, nullptr, NAUTH * HID);

    run_gat(xa, NAUTH,  xp, NPAPER, l, 0, w_src,  w_dst,  outp);  // writes a->p
    run_gat(xp, NPAPER, xp, NPAPER, l, 1, c_src,  c_dst,  outp);  // cites p->p
    run_gat(xp, NPAPER, xa, NAUTH,  l, 2, wbk_src, wbk_dst, outa); // written_by p->a

    float* dp = (l == 1) ? (float*)d_out : xp;
    float* da = (l == 1) ? ((float*)d_out + (size_t)NPAPER * HID) : xa;
    k_elu<<<cdiv((long)NPAPER * HID, 256), 256, 0, stream>>>(outp, dp, NPAPER * HID);
    k_elu<<<cdiv((long)NAUTH * HID, 256), 256, 0, stream>>>(outa, da, NAUTH * HID);
  }
}
